// TransformerDecoder_11836929868587
// MI455X (gfx1250) — compile-verified, matrix operand order checked by reading, not run on hardware
//
#include <hip/hip_runtime.h>
#include <hip/hip_bf16.h>

// ---------------------------------------------------------------------------
// Types / helpers for CDNA5 WMMA (gfx1250, wave32)
// ---------------------------------------------------------------------------
typedef __attribute__((ext_vector_type(16))) __bf16 v16bf;
typedef __attribute__((ext_vector_type(8)))  float  v8f;

union FragU { v16bf v; unsigned u[8]; };

__device__ inline v8f v8f_zero() {
    v8f z;
#pragma unroll
    for (int i = 0; i < 8; ++i) z[i] = 0.0f;
    return z;
}

__device__ inline v8f wmma_bf16(v16bf a, v16bf b, v8f c) {
    // D = A(16x32 bf16) x B(32x16 bf16) + C(16x16 f32)
    return __builtin_amdgcn_wmma_f32_16x16x32_bf16(
        /*neg_a=*/false, a, /*neg_b=*/false, b,
        /*c_mod=*/(short)0, c, /*reuse_a=*/false, /*reuse_b=*/false);
}

// 16-bit A-matrix (and K^T-style B) fragment load, row-major source.
// ISA layout: lanes 0-15 row M=lane, VGPR i holds K pair {2i (+8 if i>=4)};
// lanes 16-31 add +8 to K. Packed pairs -> aligned dword loads.
__device__ inline v16bf load_frag_rowmajor(const __hip_bfloat16* base, int rowStride) {
    const int lane = threadIdx.x & 31;
    const int row  = lane & 15;
    const int hi8  = (lane >> 4) << 3;           // 0 or 8
    FragU f;
#pragma unroll
    for (int i = 0; i < 8; ++i) {
        const int k0 = 2 * i + ((i >= 4) ? 8 : 0) + hi8;
        f.u[i] = *reinterpret_cast<const unsigned*>(base + (size_t)row * rowStride + k0);
    }
    return f.v;
}

// B fragment from a K-major (row-major [K][N]) source: column n = lane&15,
// k pairs are strided -> gather two u16 per dword.
__device__ inline v16bf load_frag_colmajor(const unsigned short* base, int rowStride) {
    const int lane = threadIdx.x & 31;
    const int n    = lane & 15;
    const int hi8  = (lane >> 4) << 3;
    FragU f;
#pragma unroll
    for (int i = 0; i < 8; ++i) {
        const int k0 = 2 * i + ((i >= 4) ? 8 : 0) + hi8;
        unsigned lo = base[(size_t)k0 * rowStride + n];
        unsigned hh = base[(size_t)(k0 + 1) * rowStride + n];
        f.u[i] = lo | (hh << 16);
    }
    return f.v;
}

// ---------------------------------------------------------------------------
// CDNA5 async global->LDS copy (ASYNCcnt-tracked, see CDNA5 ISA)
// VDST = per-lane LDS byte address (flat-shared addr low 32 bits), VADDR = 64b.
// ---------------------------------------------------------------------------
__device__ inline void async_copy_b128(unsigned ldsAddr, const void* gaddr) {
    asm volatile("global_load_async_to_lds_b128 %0, %1, off"
                 :: "v"(ldsAddr), "v"(gaddr)
                 : "memory");
}

__device__ inline void wait_asynccnt0() {
    asm volatile("s_wait_asynccnt 0x0" ::: "memory");
}

// ---------------------------------------------------------------------------
// Weight fp32 [K][N] -> bf16 transposed [N][K]
// ---------------------------------------------------------------------------
__global__ __launch_bounds__(256) void transpose_convert_kernel(
        const float* __restrict__ src, __hip_bfloat16* __restrict__ dst, int K, int N) {
    size_t idx = (size_t)blockIdx.x * 256 + threadIdx.x;
    if (idx >= (size_t)K * N) return;
    int kk = (int)(idx / N);
    int nn = (int)(idx % N);
    dst[(size_t)nn * K + kk] = __float2bfloat16(src[idx]);
}

// ---------------------------------------------------------------------------
// RMSNorm: f32 input row -> OutT (bf16 or f32) * weight
// ---------------------------------------------------------------------------
__device__ inline void store_out(float* p, float v)          { *p = v; }
__device__ inline void store_out(__hip_bfloat16* p, float v) { *p = __float2bfloat16(v); }

template <typename OutT>
__global__ __launch_bounds__(256) void rmsnorm_kernel(
        const float* __restrict__ x, const float* __restrict__ w,
        OutT* __restrict__ out, int ncols) {
    const int row = blockIdx.x;
    const float* xr = x + (size_t)row * ncols;
    float ss = 0.0f;
    for (int c = threadIdx.x; c < ncols; c += 256) {
        float v = xr[c];
        ss += v * v;
    }
#pragma unroll
    for (int off = 16; off; off >>= 1) ss += __shfl_xor(ss, off);
    __shared__ float red[8];
    __shared__ float srn;
    if ((threadIdx.x & 31) == 0) red[threadIdx.x >> 5] = ss;
    __syncthreads();
    if (threadIdx.x == 0) {
        float t = 0.0f;
#pragma unroll
        for (int i = 0; i < 8; ++i) t += red[i];
        srn = rsqrtf(t / (float)ncols + 1e-6f);
    }
    __syncthreads();
    const float rn = srn;
    for (int c = threadIdx.x; c < ncols; c += 256)
        store_out(out + (size_t)row * ncols + c, xr[c] * rn * w[c]);
}

// ---------------------------------------------------------------------------
// WMMA GEMM: out[M,N] = A[M,K](bf16) * W[K,N], W given transposed Wt[N][K].
// Block 256 thr = 8 waves, tile 128x128, K-step 32.
// Double-buffered async global->LDS pipeline:
//   iter kt: s_wait_asynccnt(tile kt arrived) -> barrier ->
//            issue async fill of tile kt+1 -> 8 WMMAs on tile kt.
// LDS row stride 40 bf16 (80B): 16B-aligned for async B128 stores and
// conflict-free fragment reads (20*row mod 64 injective over 16 rows).
// ADD_RES==1: accumulate into fp32 outF (residual add); else store bf16 outB.
// ---------------------------------------------------------------------------
#define GEMM_LDS_STRIDE 40

template <int ADD_RES>
__global__ __launch_bounds__(256) void gemm_bf16_kernel(
        const __hip_bfloat16* __restrict__ A, const __hip_bfloat16* __restrict__ Wt,
        __hip_bfloat16* __restrict__ outB, float* __restrict__ outF,
        int M, int N, int K) {
    __shared__ __align__(16) __hip_bfloat16 A_s[2][128 * GEMM_LDS_STRIDE];
    __shared__ __align__(16) __hip_bfloat16 B_s[2][128 * GEMM_LDS_STRIDE];

    const int bm   = blockIdx.y * 128;
    const int bn   = blockIdx.x * 128;
    const int tid  = threadIdx.x;
    const int wave = tid >> 5;
    const int wm   = wave >> 1;   // 0..3 -> 32-row slice
    const int wn   = wave & 1;    // 0..1 -> 64-col slice
    const int lane = tid & 31;
    const int n15  = lane & 15;
    const int hi   = lane >> 4;

    v8f acc[2][4];
#pragma unroll
    for (int mi = 0; mi < 2; ++mi)
#pragma unroll
        for (int ni = 0; ni < 4; ++ni) acc[mi][ni] = v8f_zero();

    const int lrow = tid >> 1;            // 0..127
    const int lseg = (tid & 1) * 16;      // element offset 0/16

    // per-thread source/dest cursors for the async fill (2x B128 for A, 2x for B)
    const __hip_bfloat16* gA = A  + (size_t)(bm + lrow) * K + lseg;
    const __hip_bfloat16* gB = Wt + (size_t)(bn + lrow) * K + lseg;
    unsigned ldsA[2], ldsB[2];
#pragma unroll
    for (int s = 0; s < 2; ++s) {
        ldsA[s] = (unsigned)(uintptr_t)(&A_s[s][lrow * GEMM_LDS_STRIDE + lseg]);
        ldsB[s] = (unsigned)(uintptr_t)(&B_s[s][lrow * GEMM_LDS_STRIDE + lseg]);
    }

    const int T = K >> 5;   // number of 32-wide K tiles

    // prologue: fill buffer 0 with tile 0
    async_copy_b128(ldsA[0],      gA);
    async_copy_b128(ldsA[0] + 16, gA + 8);
    async_copy_b128(ldsB[0],      gB);
    async_copy_b128(ldsB[0] + 16, gB + 8);

    for (int kt = 0; kt < T; ++kt) {
        const int cur = kt & 1;
        wait_asynccnt0();      // tile kt landed in LDS (this wave's portion)
        __syncthreads();       // all waves' portions landed

        if (kt + 1 < T) {      // overlap: async-fill next tile while computing
            const int nxt = cur ^ 1;
            const __hip_bfloat16* nA = gA + (size_t)(kt + 1) * 32;
            const __hip_bfloat16* nB = gB + (size_t)(kt + 1) * 32;
            async_copy_b128(ldsA[nxt],      nA);
            async_copy_b128(ldsA[nxt] + 16, nA + 8);
            async_copy_b128(ldsB[nxt],      nB);
            async_copy_b128(ldsB[nxt] + 16, nB + 8);
        }

        v16bf af[2], bfr[4];
#pragma unroll
        for (int mi = 0; mi < 2; ++mi)
            af[mi] = load_frag_rowmajor(
                &A_s[cur][(wm * 32 + mi * 16) * GEMM_LDS_STRIDE], GEMM_LDS_STRIDE);
#pragma unroll
        for (int ni = 0; ni < 4; ++ni)
            bfr[ni] = load_frag_rowmajor(
                &B_s[cur][(wn * 64 + ni * 16) * GEMM_LDS_STRIDE], GEMM_LDS_STRIDE);
#pragma unroll
        for (int mi = 0; mi < 2; ++mi)
#pragma unroll
            for (int ni = 0; ni < 4; ++ni)
                acc[mi][ni] = wmma_bf16(af[mi], bfr[ni], acc[mi][ni]);
        // no trailing barrier: next overwrite of this buffer is gated by the
        // barrier at the top of iteration kt+1 (all waves past compute kt).
    }

#pragma unroll
    for (int mi = 0; mi < 2; ++mi)
#pragma unroll
        for (int ni = 0; ni < 4; ++ni)
#pragma unroll
            for (int r = 0; r < 8; ++r) {
                const int row = bm + wm * 32 + mi * 16 + r + hi * 8;
                const int col = bn + wn * 64 + ni * 16 + n15;
                if (ADD_RES) outF[(size_t)row * N + col] += acc[mi][ni][r];
                else         outB[(size_t)row * N + col] = __float2bfloat16(acc[mi][ni][r]);
            }
}

// ---------------------------------------------------------------------------
// RoPE on q,k (bf16, [B,S,NH,hd]) from raw timestamps
// ---------------------------------------------------------------------------
__global__ __launch_bounds__(256) void rope_kernel(
        __hip_bfloat16* __restrict__ q, __hip_bfloat16* __restrict__ k,
        const float* __restrict__ ts, int B, int S, int NH, int hd) {
    const int pairs = hd >> 1;
    size_t idx = (size_t)blockIdx.x * 256 + threadIdx.x;
    if (idx >= (size_t)B * S * NH * pairs) return;
    int p = (int)(idx % pairs); size_t t = idx / pairs;
    int hh = (int)(t % NH); t /= NH;
    int s = (int)(t % S);  int b = (int)(t / S);

    const float tsv = ts[(size_t)b * S + s];
    const float inv = __expf(-(2.0f * p / (float)hd) * __logf(10000.0f));
    float sn, cs;
    __sincosf(tsv * inv, &sn, &cs);
    const size_t o = (((size_t)b * S + s) * NH + hh) * hd + 2 * p;

    float x0 = __bfloat162float(q[o]), x1 = __bfloat162float(q[o + 1]);
    q[o]     = __float2bfloat16(x0 * cs - x1 * sn);
    q[o + 1] = __float2bfloat16(x0 * sn + x1 * cs);
    x0 = __bfloat162float(k[o]); x1 = __bfloat162float(k[o + 1]);
    k[o]     = __float2bfloat16(x0 * cs - x1 * sn);
    k[o + 1] = __float2bfloat16(x0 * sn + x1 * cs);
}

// ---------------------------------------------------------------------------
// Flash attention, WMMA bf16. Block = (b, h, 64 q rows); 4 waves x 16 q rows.
// ---------------------------------------------------------------------------
__global__ __launch_bounds__(128) void attention_kernel(
        const __hip_bfloat16* __restrict__ q, const __hip_bfloat16* __restrict__ k,
        const __hip_bfloat16* __restrict__ v, const unsigned char* __restrict__ exist,
        __hip_bfloat16* __restrict__ out, int B, int S, int NH, int hd) {
    const int D    = NH * hd;
    const int wave = threadIdx.x >> 5;
    const int lane = threadIdx.x & 31;
    const int n15  = lane & 15;
    const int hi   = lane >> 4;
    const int b = blockIdx.z, h = blockIdx.y;
    const int qbase = blockIdx.x * 64 + wave * 16;

    const __hip_bfloat16* Qh = q + (size_t)b * S * D + (size_t)h * hd;
    const __hip_bfloat16* Kh = k + (size_t)b * S * D + (size_t)h * hd;
    const unsigned short* Vh = reinterpret_cast<const unsigned short*>(
                                   v + (size_t)b * S * D + (size_t)h * hd);
    const unsigned char* exb = exist + (size_t)b * S;

    const v16bf a0 = load_frag_rowmajor(Qh + (size_t)qbase * D, D);       // hd 0..31
    const v16bf a1 = load_frag_rowmajor(Qh + (size_t)qbase * D + 32, D);  // hd 32..63

    float mrow[8], lrow[8];
    v8f O[4];
#pragma unroll
    for (int r = 0; r < 8; ++r) { mrow[r] = -3e38f; lrow[r] = 0.0f; }
#pragma unroll
    for (int t = 0; t < 4; ++t) O[t] = v8f_zero();

    __shared__ unsigned short P_s[4 * 16 * 34];   // per-wave P tile (16x32 bf16, padded)
    unsigned short* Pw = P_s + wave * 16 * 34;

    const float scale = 0.125f;                   // 1/sqrt(64)
    const int ktend = (qbase + 16 + 31) >> 5;     // causal: only key tiles <= q range

    for (int kt = 0; kt < ktend; ++kt) {
        const int ks = kt * 32;
        // K^T B-fragments load contiguously along hd (row-major per key)
        const v16bf b00 = load_frag_rowmajor(Kh + (size_t)ks * D, D);
        const v16bf b01 = load_frag_rowmajor(Kh + (size_t)ks * D + 32, D);
        const v16bf b10 = load_frag_rowmajor(Kh + (size_t)(ks + 16) * D, D);
        const v16bf b11 = load_frag_rowmajor(Kh + (size_t)(ks + 16) * D + 32, D);

        v8f s0 = v8f_zero(), s1 = v8f_zero();
        s0 = wmma_bf16(a0, b00, s0); s0 = wmma_bf16(a1, b01, s0);
        s1 = wmma_bf16(a0, b10, s1); s1 = wmma_bf16(a1, b11, s1);

        const int key0 = ks + n15, key1 = ks + 16 + n15;
        const bool ex0 = exb[key0] != 0, ex1 = exb[key1] != 0;

#pragma unroll
        for (int r = 0; r < 8; ++r) {
            const int m    = r + hi * 8;
            const int qrow = qbase + m;
            float x0 = s0[r] * scale, x1 = s1[r] * scale;
            if (key0 > qrow || (!ex0 && key0 != qrow)) x0 = -3e38f;
            if (key1 > qrow || (!ex1 && key1 != qrow)) x1 = -3e38f;

            float vmax = fmaxf(x0, x1);
#pragma unroll
            for (int off = 8; off; off >>= 1) vmax = fmaxf(vmax, __shfl_xor(vmax, off));
            const float mo = mrow[r];
            const float mn = fmaxf(mo, vmax);
            const float alpha = __expf(mo - mn);
            const float p0 = __expf(x0 - mn), p1 = __expf(x1 - mn);
            float rs = p0 + p1;
#pragma unroll
            for (int off = 8; off; off >>= 1) rs += __shfl_xor(rs, off);
            mrow[r] = mn;
            lrow[r] = lrow[r] * alpha + rs;
#pragma unroll
            for (int t = 0; t < 4; ++t) O[t][r] *= alpha;

            __hip_bfloat16 hp0 = __float2bfloat16(p0);
            __hip_bfloat16 hp1 = __float2bfloat16(p1);
            Pw[m * 34 + n15]      = *reinterpret_cast<unsigned short*>(&hp0);
            Pw[m * 34 + 16 + n15] = *reinterpret_cast<unsigned short*>(&hp1);
        }
        // C-layout -> A-layout through LDS, then P(16x32) x V(32x64)
        const v16bf pf = load_frag_rowmajor(
            reinterpret_cast<const __hip_bfloat16*>(Pw), 34);
#pragma unroll
        for (int t = 0; t < 4; ++t) {
            const v16bf vf = load_frag_colmajor(Vh + (size_t)ks * D + t * 16, D);
            O[t] = wmma_bf16(pf, vf, O[t]);
        }
    }

    __hip_bfloat16* Oh = out + (size_t)b * S * D + (size_t)h * hd;
#pragma unroll
    for (int r = 0; r < 8; ++r) {
        const float inv = 1.0f / lrow[r];
        const int m = r + hi * 8;
#pragma unroll
        for (int t = 0; t < 4; ++t)
            Oh[(size_t)(qbase + m) * D + t * 16 + n15] = __float2bfloat16(O[t][r] * inv);
    }
}

// ---------------------------------------------------------------------------
// SwiGLU elementwise: hb = silu(g1) * g3
// ---------------------------------------------------------------------------
__global__ __launch_bounds__(256) void silu_mul_kernel(
        const __hip_bfloat16* __restrict__ g1, const __hip_bfloat16* __restrict__ g3,
        __hip_bfloat16* __restrict__ hb, size_t n) {
    size_t i = (size_t)blockIdx.x * 256 + threadIdx.x;
    if (i >= n) return;
    float x = __bfloat162float(g1[i]);
    float s = x / (1.0f + __expf(-x));
    hb[i] = __float2bfloat16(s * __bfloat162float(g3[i]));
}

// ---------------------------------------------------------------------------
// Host orchestration
// ---------------------------------------------------------------------------
extern "C" void kernel_launch(void* const* d_in, const int* in_sizes, int n_in,
                              void* d_out, int out_size, void* d_ws, size_t ws_size,
                              hipStream_t stream) {
    (void)in_sizes; (void)n_in; (void)out_size; (void)ws_size;
    const int B = 2, S = 1024, D = 1024, L = 6, HID = 2816, NH = 16, HD = 64;
    const int M = B * S;

    const float* hs  = (const float*)d_in[0];
    const float* ts  = (const float*)d_in[1];
    const unsigned char* ex = (const unsigned char*)d_in[2];
    const float* Wq  = (const float*)d_in[3];
    const float* Wk  = (const float*)d_in[4];
    const float* Wv  = (const float*)d_in[5];
    const float* Wo  = (const float*)d_in[6];
    const float* W1  = (const float*)d_in[7];
    const float* W2  = (const float*)d_in[8];
    const float* W3  = (const float*)d_in[9];
    const float* anw = (const float*)d_in[10];
    const float* ffw = (const float*)d_in[11];
    const float* fnw = (const float*)d_in[12];

    char* ws = (char*)d_ws;
    size_t off = 0;
    auto alloc = [&](size_t bytes) -> void* {
        void* p = ws + off;
        off += (bytes + 255) & ~(size_t)255;
        return p;
    };
    float*           h    = (float*)alloc((size_t)M * D * 4);
    __hip_bfloat16*  xn   = (__hip_bfloat16*)alloc((size_t)M * D * 2);
    __hip_bfloat16*  qb   = (__hip_bfloat16*)alloc((size_t)M * D * 2);
    __hip_bfloat16*  kb   = (__hip_bfloat16*)alloc((size_t)M * D * 2);
    __hip_bfloat16*  vb   = (__hip_bfloat16*)alloc((size_t)M * D * 2);
    __hip_bfloat16*  attn = (__hip_bfloat16*)alloc((size_t)M * D * 2);
    __hip_bfloat16*  g1   = (__hip_bfloat16*)alloc((size_t)M * HID * 2);
    __hip_bfloat16*  g3   = (__hip_bfloat16*)alloc((size_t)M * HID * 2);
    __hip_bfloat16*  hb   = (__hip_bfloat16*)alloc((size_t)M * HID * 2);
    __hip_bfloat16*  Wqt  = (__hip_bfloat16*)alloc((size_t)L * D * D * 2);
    __hip_bfloat16*  Wkt  = (__hip_bfloat16*)alloc((size_t)L * D * D * 2);
    __hip_bfloat16*  Wvt  = (__hip_bfloat16*)alloc((size_t)L * D * D * 2);
    __hip_bfloat16*  Wot  = (__hip_bfloat16*)alloc((size_t)L * D * D * 2);
    __hip_bfloat16*  W1t  = (__hip_bfloat16*)alloc((size_t)L * D * HID * 2);
    __hip_bfloat16*  W3t  = (__hip_bfloat16*)alloc((size_t)L * D * HID * 2);
    __hip_bfloat16*  W2t  = (__hip_bfloat16*)alloc((size_t)L * HID * D * 2);

    // residual stream copy (inputs must not be mutated)
    hipMemcpyAsync(h, hs, (size_t)M * D * 4, hipMemcpyDeviceToDevice, stream);

    // weights -> bf16 transposed [N][K]
    const int DD = D * D, DH = D * HID;
    const int gDD = DD / 256, gDH = (DH + 255) / 256;
    for (int l = 0; l < L; ++l) {
        transpose_convert_kernel<<<gDD, 256, 0, stream>>>(Wq + (size_t)l * DD, Wqt + (size_t)l * DD, D, D);
        transpose_convert_kernel<<<gDD, 256, 0, stream>>>(Wk + (size_t)l * DD, Wkt + (size_t)l * DD, D, D);
        transpose_convert_kernel<<<gDD, 256, 0, stream>>>(Wv + (size_t)l * DD, Wvt + (size_t)l * DD, D, D);
        transpose_convert_kernel<<<gDD, 256, 0, stream>>>(Wo + (size_t)l * DD, Wot + (size_t)l * DD, D, D);
        transpose_convert_kernel<<<gDH, 256, 0, stream>>>(W1 + (size_t)l * DH, W1t + (size_t)l * DH, D, HID);
        transpose_convert_kernel<<<gDH, 256, 0, stream>>>(W3 + (size_t)l * DH, W3t + (size_t)l * DH, D, HID);
        transpose_convert_kernel<<<gDH, 256, 0, stream>>>(W2 + (size_t)l * DH, W2t + (size_t)l * DH, HID, D);
    }

    const dim3 gemmDDgrid(D / 128, M / 128);        // (8,16)
    const dim3 gemmDHgrid(HID / 128, M / 128);      // (22,16)
    const dim3 attnGrid(S / 64, NH, B);
    const int ropeBlocks = (M * NH * (HD / 2) + 255) / 256;
    const int siluBlocks = (int)(((size_t)M * HID + 255) / 256);

    for (int l = 0; l < L; ++l) {
        rmsnorm_kernel<__hip_bfloat16><<<M, 256, 0, stream>>>(h, anw + (size_t)l * D, xn, D);
        gemm_bf16_kernel<0><<<gemmDDgrid, 256, 0, stream>>>(xn, Wqt + (size_t)l * DD, qb, nullptr, M, D, D);
        gemm_bf16_kernel<0><<<gemmDDgrid, 256, 0, stream>>>(xn, Wkt + (size_t)l * DD, kb, nullptr, M, D, D);
        gemm_bf16_kernel<0><<<gemmDDgrid, 256, 0, stream>>>(xn, Wvt + (size_t)l * DD, vb, nullptr, M, D, D);
        rope_kernel<<<ropeBlocks, 256, 0, stream>>>(qb, kb, ts, B, S, NH, HD);
        attention_kernel<<<attnGrid, 128, 0, stream>>>(qb, kb, vb, ex, attn, B, S, NH, HD);
        gemm_bf16_kernel<1><<<gemmDDgrid, 256, 0, stream>>>(attn, Wot + (size_t)l * DD, nullptr, h, M, D, D);
        rmsnorm_kernel<__hip_bfloat16><<<M, 256, 0, stream>>>(h, ffw + (size_t)l * D, xn, D);
        gemm_bf16_kernel<0><<<gemmDHgrid, 256, 0, stream>>>(xn, W1t + (size_t)l * DH, g1, nullptr, M, HID, D);
        gemm_bf16_kernel<0><<<gemmDHgrid, 256, 0, stream>>>(xn, W3t + (size_t)l * DH, g3, nullptr, M, HID, D);
        silu_mul_kernel<<<siluBlocks, 256, 0, stream>>>(g1, g3, hb, (size_t)M * HID);
        gemm_bf16_kernel<1><<<gemmDDgrid, 256, 0, stream>>>(hb, W2t + (size_t)l * DH, nullptr, h, M, D, HID);
    }
    rmsnorm_kernel<float><<<M, 256, 0, stream>>>(h, fnw, (float*)d_out, D);
}
